// HexConvResLayer_18339510353957
// MI455X (gfx1250) — compile-verified, matrix operand order checked by reading, not run on hardware
//
#include <hip/hip_runtime.h>

// ---------------------------------------------------------------------------
// HexConv-as-GEMM for MI455X (gfx1250): D = leakyrelu(gather(x) @ W + b) + x
//   M = 256*165 = 42240, K = 7*512 = 3584, N = 512
//   Compute-bound (AI ~ 860 flop/byte) -> v_wmma_f32_16x16x32_f16 path.
//   WG tile 128Mx128N, 8 waves of 32Mx64N each -> 8 WMMA per K-step per wave.
// ---------------------------------------------------------------------------

typedef __attribute__((ext_vector_type(16))) _Float16     v16h;
typedef __attribute__((ext_vector_type(8)))  float        v8f;
typedef __attribute__((ext_vector_type(4)))  unsigned int u32x4;
typedef __attribute__((ext_vector_type(4)))  float        f32x4;

#define BATCH 256
#define NHEX  165
#define CCH   512
#define KDIM  3584               // 7*CCH
#define MROWS (BATCH * NHEX)     // 42240 = 330 * 128
#define MT    128
#define NT    128
#define KT    32
#define PITCH 40                 // halves per LDS row (32 data + 8 pad) -> conflict-free b128

union ABfrag { struct { u32x4 lo, hi; } q; v16h h; };
union H8     { _Float16 h[8]; u32x4 u; };

// ---- one-time W[f32][K][N] -> Wt[f16][N][K] convert+transpose ----
__global__ __launch_bounds__(256) void hexconv_wcast_kernel(
    const float* __restrict__ W, _Float16* __restrict__ Wt) {
  const int idx = blockIdx.x * 256 + threadIdx.x;      // over N*K = 512*3584
  const int n   = idx / KDIM;
  const int k   = idx - n * KDIM;
  Wt[idx] = (_Float16)W[(size_t)k * CCH + n];
}

// ---- main fused gather + GEMM + bias + leakyrelu + residual ----
__global__ __launch_bounds__(256) void hexconv_gemm_kernel(
    const float* __restrict__ x, const _Float16* __restrict__ Wt,
    const float* __restrict__ bias, float* __restrict__ out) {
  __shared__ _Float16 ldsA[MT * PITCH];   // 10 KB
  __shared__ _Float16 ldsB[NT * PITCH];   // 10 KB

  const int tid    = threadIdx.x;
  const int lane   = tid & 31;
  const int wave   = tid >> 5;            // 8 waves
  const int wm     = wave >> 1;           // 0..3 -> 32-row slice
  const int wn     = wave & 1;            // 0..1 -> 64-col slice
  const int laneN  = lane & 15;           // M (A) or N (B/C) index
  const int laneHi = lane >> 4;           // 0 or 1
  const int kb     = laneHi << 3;         // lane-group K sub-base (0 or 8)

  const int mBase = blockIdx.x * MT;
  const int n0    = blockIdx.y * NT;

  // ---------- A-loader state: this thread stages row (mBase+mLocal), 16 K's ----------
  const int mLocal = tid >> 1;            // 0..127
  const int kq     = (tid & 1) << 4;      // 0 or 16 within 32-wide K step

  const float* aptr[7];
  {
    const int R  = mBase + mLocal;
    const int b_ = R / NHEX;
    const int n_ = R - b_ * NHEX;
    const int gy = n_ / 15;
    const int gx = n_ - gy * 15;
    const int py0 = gy + 1;
    const int h   = py0 * 17 + gx + 1;      // index into padded 13x17 grid
    const bool evenRow = (py0 & 1) == 0;
    const int off0[7] = {-17, -16, -1, 0, 1, 17, 18};
    const int off1[7] = {-18, -17, -1, 0, 1, 16, 17};
    const float* xb = x + (size_t)b_ * NHEX * CCH;
#pragma unroll
    for (int t = 0; t < 7; ++t) {
      const int p  = h + (evenRow ? off0[t] : off1[t]);
      const int py = p / 17;
      const int px = p - py * 17;
      const bool valid = (py >= 1) && (py <= 11) && (px >= 1) && (px <= 15);
      aptr[t] = valid ? (xb + (size_t)((py - 1) * 15 + (px - 1)) * CCH)
                      : (const float*)0;   // zero-padded neighbor
    }
  }

  // ---------- B-loader state: this thread stages Wt row (n0 + tid>>1), 16 K's ----------
  const int nRow = tid >> 1;              // 0..127
  const int koff = (tid & 1) << 4;        // 0 or 16 halves
  const _Float16* bsrc = Wt + (size_t)(n0 + nRow) * KDIM + koff;

  const v8f zero8 = {0.f, 0.f, 0.f, 0.f, 0.f, 0.f, 0.f, 0.f};
  v8f acc[2][4] = {{zero8, zero8, zero8, zero8},
                   {zero8, zero8, zero8, zero8}};

  // ---------- K loop: 7 neighbors x 16 steps of K=32 ----------
#pragma unroll
  for (int t = 0; t < 7; ++t) {
    const float* ap = aptr[t];
    for (int ks = 0; ks < 16; ++ks) {
      const int k0 = (t * 16 + ks) * KT;

      // issue global loads for this step first (overlap with barrier)
      f32x4 a0 = {0.f, 0.f, 0.f, 0.f};
      f32x4 a1 = {0.f, 0.f, 0.f, 0.f};
      f32x4 a2 = {0.f, 0.f, 0.f, 0.f};
      f32x4 a3 = {0.f, 0.f, 0.f, 0.f};
      if (ap) {
        const float* p = ap + (k0 & (CCH - 1)) + kq;
        a0 = *(const f32x4*)p;
        a1 = *(const f32x4*)(p + 4);
        a2 = *(const f32x4*)(p + 8);
        a3 = *(const f32x4*)(p + 12);
      }
      const u32x4 b0 = *(const u32x4*)(bsrc + k0);
      const u32x4 b1 = *(const u32x4*)(bsrc + k0 + 8);
      if (k0 + KT < KDIM)
        __builtin_prefetch((const void*)(bsrc + k0 + KT), 0, 1);  // global_prefetch

      __syncthreads();                    // previous step's LDS reads done

      H8 p0, p1;
#pragma unroll
      for (int i = 0; i < 4; ++i) {
        p0.h[i]     = (_Float16)a0[i];
        p0.h[4 + i] = (_Float16)a1[i];
        p1.h[i]     = (_Float16)a2[i];
        p1.h[4 + i] = (_Float16)a3[i];
      }
      *(u32x4*)&ldsA[mLocal * PITCH + kq]       = p0.u;
      *(u32x4*)&ldsA[mLocal * PITCH + kq + 8]   = p1.u;
      *(u32x4*)&ldsB[nRow * PITCH + koff]       = b0;
      *(u32x4*)&ldsB[nRow * PITCH + koff + 8]   = b1;

      __syncthreads();                    // tiles visible

      // A fragments: 16x32 f16 (lanes 0-15: K 0..7/16..23, lanes 16-31: K 8..15/24..31)
      ABfrag a[2];
#pragma unroll
      for (int mf = 0; mf < 2; ++mf) {
        const _Float16* pa = &ldsA[(wm * 32 + mf * 16 + laneN) * PITCH + kb];
        a[mf].q.lo = *(const u32x4*)pa;
        a[mf].q.hi = *(const u32x4*)(pa + 16);
      }

#pragma unroll
      for (int f = 0; f < 4; ++f) {
        ABfrag bb;
        const _Float16* pb = &ldsB[(wn * 64 + f * 16 + laneN) * PITCH + kb];
        bb.q.lo = *(const u32x4*)pb;
        bb.q.hi = *(const u32x4*)(pb + 16);
#pragma unroll
        for (int mf = 0; mf < 2; ++mf) {
          acc[mf][f] = __builtin_amdgcn_wmma_f32_16x16x32_f16(
              false, a[mf].h, false, bb.h, (short)0, acc[mf][f], false, false);
        }
      }
    }
  }

  // ---------- epilogue: bias + leaky relu + residual ----------
#pragma unroll
  for (int f = 0; f < 4; ++f) {
    const int col = n0 + wn * 64 + f * 16 + laneN;
    const float bv = bias[col];
#pragma unroll
    for (int mf = 0; mf < 2; ++mf) {
#pragma unroll
      for (int r = 0; r < 8; ++r) {
        const int R = mBase + wm * 32 + mf * 16 + laneHi * 8 + r;  // C: vgpr r -> M=r (+8 hi)
        float yv = acc[mf][f][r] + bv;
        yv = (yv > 0.f) ? yv : 0.01f * yv;
        const size_t o = (size_t)R * CCH + col;
        out[o] = x[o] + yv;
      }
    }
  }
}

extern "C" void kernel_launch(void* const* d_in, const int* in_sizes, int n_in,
                              void* d_out, int out_size, void* d_ws, size_t ws_size,
                              hipStream_t stream) {
  const float* x    = (const float*)d_in[0];   // [256,165,512] f32
  const float* W    = (const float*)d_in[1];   // [3584,512]    f32
  const float* bias = (const float*)d_in[2];   // [512]         f32
  float*       out  = (float*)d_out;           // [256,165,512] f32
  _Float16*    Wt   = (_Float16*)d_ws;         // [512,3584]    f16 (3.67 MB)

  hexconv_wcast_kernel<<<(CCH * KDIM) / 256, 256, 0, stream>>>(W, Wt);

  dim3 grid(MROWS / MT, CCH / NT);             // 330 x 4
  hexconv_gemm_kernel<<<grid, 256, 0, stream>>>(x, Wt, bias, out);
}
